// SQBnBLinear_24489903521828
// MI455X (gfx1250) — 8-bit matrix operand order repaired and checked by reading, compiled, not run on hardware
//
#include <hip/hip_runtime.h>
#include <hip/hip_bf16.h>

typedef int v8i __attribute__((ext_vector_type(8)));

__global__ __launch_bounds__(256) void quant_rows_kernel(
    const float* __restrict__ src, const float* __restrict__ scales,
    signed char* __restrict__ q, float* __restrict__ amax_out,
    int ncols, int mode) {
  const int row  = blockIdx.x;
  const int tid  = threadIdx.x;
  const int lane = tid & 31;
  const int wv   = tid >> 5;

  const float* rowp = src + (size_t)row * ncols;

  float vals[16];
  float m = 0.0f;
  int nIter = ncols >> 10;
  #pragma unroll
  for (int it = 0; it < 4; ++it) {
    if (it >= nIter) break;
    int idx = (it << 10) + (tid << 2);
    float4 v = *reinterpret_cast<const float4*>(rowp + idx);
    float4 s = *reinterpret_cast<const float4*>(scales + idx);
    float a, b, c, d;
    if (mode == 0) { a = v.x * s.x; b = v.y * s.y; c = v.z * s.z; d = v.w * s.w; }
    else           { a = v.x / s.x; b = v.y / s.y; c = v.z / s.z; d = v.w / s.w; }
    vals[it * 4 + 0] = a; vals[it * 4 + 1] = b;
    vals[it * 4 + 2] = c; vals[it * 4 + 3] = d;
    m = fmaxf(m, fmaxf(fmaxf(fabsf(a), fabsf(b)), fmaxf(fabsf(c), fabsf(d))));
  }

  #pragma unroll
  for (int off = 16; off > 0; off >>= 1)
    m = fmaxf(m, __shfl_xor(m, off, 32));
  __shared__ float red[8];
  if (lane == 0) red[wv] = m;
  __syncthreads();
  m = red[0];
  #pragma unroll
  for (int i = 1; i < 8; ++i) m = fmaxf(m, red[i]);
  m = fmaxf(m, 1e-8f);

  if (tid == 0) amax_out[row] = m;

  const float inv = 127.0f / m;
  #pragma unroll
  for (int it = 0; it < 4; ++it) {
    if (it >= nIter) break;
    int idx = (it << 10) + (tid << 2);
    int packed = 0;
    #pragma unroll
    for (int j = 0; j < 4; ++j) {
      float qf = rintf(vals[it * 4 + j] * inv);
      qf = fminf(fmaxf(qf, -127.0f), 127.0f);
      packed |= (((int)qf) & 0xFF) << (8 * j);
    }
    *reinterpret_cast<int*>(q + (size_t)row * ncols + idx) = packed;
  }
}

__global__ __launch_bounds__(256) void bias_fold_kernel(
    const float* __restrict__ bias, const float* __restrict__ scales,
    float* __restrict__ bsq, int n) {
  int i = blockIdx.x * 256 + threadIdx.x;
  if (i < n) bsq[i] = bias[i] / scales[i];
}

#define LDS_STRIDE 80

__global__ __launch_bounds__(256, 1) void sq_int8_gemm_kernel(
    const signed char* __restrict__ qA,
    const signed char* __restrict__ qB,
    const float* __restrict__ ax,
    const float* __restrict__ aw,
    const float* __restrict__ bsq,
    float* __restrict__ out,
    int M, int N, int K) {
  __shared__ alignas(16) signed char lA[128 * LDS_STRIDE];
  __shared__ alignas(16) signed char lB[256 * LDS_STRIDE];

  const int tid  = threadIdx.x;
  const int wave = tid >> 5;
  const int lane = tid & 31;
  const int lm   = lane & 15;
  const int lh   = lane >> 4;

  const int waveM = wave & 1;
  const int waveN = wave >> 1;

  const int blockM = blockIdx.y * 128;
  const int blockN = blockIdx.x * 256;

  v8i acc[4][4] = {};

  const signed char* aseg[2];
  const signed char* bseg[4];
  int aoff[2], boff[4];
  #pragma unroll
  for (int j = 0; j < 2; ++j) {
    int s = j * 256 + tid;
    int r = s >> 2, c = (s & 3) * 16;
    aseg[j] = qA + (size_t)(blockM + r) * K + c;
    aoff[j] = r * LDS_STRIDE + c;
  }
  #pragma unroll
  for (int j = 0; j < 4; ++j) {
    int s = j * 256 + tid;
    int r = s >> 2, c = (s & 3) * 16;
    bseg[j] = qB + (size_t)(blockN + r) * K + c;
    boff[j] = r * LDS_STRIDE + c;
  }

  for (int k0 = 0; k0 < K; k0 += 64) {
    __syncthreads();
    {
      int4 ra0 = *reinterpret_cast<const int4*>(aseg[0] + k0);
      int4 ra1 = *reinterpret_cast<const int4*>(aseg[1] + k0);
      int4 rb0 = *reinterpret_cast<const int4*>(bseg[0] + k0);
      int4 rb1 = *reinterpret_cast<const int4*>(bseg[1] + k0);
      int4 rb2 = *reinterpret_cast<const int4*>(bseg[2] + k0);
      int4 rb3 = *reinterpret_cast<const int4*>(bseg[3] + k0);
      *reinterpret_cast<int4*>(&lA[aoff[0]]) = ra0;
      *reinterpret_cast<int4*>(&lA[aoff[1]]) = ra1;
      *reinterpret_cast<int4*>(&lB[boff[0]]) = rb0;
      *reinterpret_cast<int4*>(&lB[boff[1]]) = rb1;
      *reinterpret_cast<int4*>(&lB[boff[2]]) = rb2;
      *reinterpret_cast<int4*>(&lB[boff[3]]) = rb3;
    }
    __syncthreads();

    if (k0 + 64 < K) {
      #pragma unroll
      for (int j = 0; j < 2; ++j)
        __builtin_prefetch(aseg[j] + k0 + 64, 0, 3);
      #pragma unroll
      for (int j = 0; j < 4; ++j)
        __builtin_prefetch(bseg[j] + k0 + 64, 0, 3);
    }

    v8i afrag[4];
    #pragma unroll
    for (int mt = 0; mt < 4; ++mt) {
      const signed char* ap =
          &lA[(waveM * 64 + mt * 16 + lm) * LDS_STRIDE + lh * 16];
      #pragma unroll
      for (int c = 0; c < 4; ++c) {
        int2 t = *reinterpret_cast<const int2*>(ap + (c >> 1) * 32 + (c & 1) * 8);
        afrag[mt][2 * c]     = t.x;
        afrag[mt][2 * c + 1] = t.y;
      }
    }

    v8i bfrag[4];
    #pragma unroll
    for (int nt = 0; nt < 4; ++nt) {
      const signed char* bp =
          &lB[(waveN * 64 + nt * 16 + lm) * LDS_STRIDE + lh * 16];
      #pragma unroll
      for (int h = 0; h < 2; ++h) {
        int4 t = *reinterpret_cast<const int4*>(bp + h * 32);
        bfrag[nt][4 * h + 0] = t.x;
        bfrag[nt][4 * h + 1] = t.y;
        bfrag[nt][4 * h + 2] = t.z;
        bfrag[nt][4 * h + 3] = t.w;
      }
    }

    #pragma unroll
    for (int mt = 0; mt < 4; ++mt)
      #pragma unroll
      for (int nt = 0; nt < 4; ++nt)
        acc[mt][nt] = __builtin_amdgcn_wmma_i32_16x16x64_iu8(
             true, afrag[mt],  true, bfrag[nt],
            acc[mt][nt],  false,  false);
  }

  const float inv127sq = 1.0f / 16129.0f;
  float awc[4], bc[4];
  #pragma unroll
  for (int nt = 0; nt < 4; ++nt) {
    const int col = blockN + waveN * 64 + nt * 16 + lm;
    awc[nt] = aw[col] * inv127sq;
    bc[nt]  = bsq[col];
  }
  #pragma unroll
  for (int mt = 0; mt < 4; ++mt) {
    float axs[8];
    #pragma unroll
    for (int r = 0; r < 8; ++r)
      axs[r] = ax[blockM + waveM * 64 + mt * 16 + r + lh * 8];
    #pragma unroll
    for (int nt = 0; nt < 4; ++nt) {
      const int col = blockN + waveN * 64 + nt * 16 + lm;
      #pragma unroll
      for (int r = 0; r < 8; ++r) {
        const int row = blockM + waveM * 64 + mt * 16 + r + lh * 8;
        out[(size_t)row * N + col] =
            (float)acc[mt][nt][r] * (axs[r] * awc[nt]) + bc[nt];
      }
    }
  }
}

static inline size_t align256(size_t v) { return (v + 255) & ~(size_t)255; }

extern "C" void kernel_launch(void* const* d_in, const int* in_sizes, int n_in,
                              void* d_out, int out_size, void* d_ws, size_t ws_size,
                              hipStream_t stream) {
  const float* x      = (const float*)d_in[0];
  const float* weight = (const float*)d_in[1];
  const float* bias   = (const float*)d_in[2];
  const float* scales = (const float*)d_in[3];

  const int D = in_sizes[3];
  const int M = in_sizes[0] / D;
  const int N = D;
  const int K = D;

  char* ws = (char*)d_ws;
  size_t off = 0;
  signed char* qx = (signed char*)(ws + off); off += align256((size_t)M * K);
  signed char* qw = (signed char*)(ws + off); off += align256((size_t)N * K);
  float* ax  = (float*)(ws + off); off += align256((size_t)M * sizeof(float));
  float* aw  = (float*)(ws + off); off += align256((size_t)N * sizeof(float));
  float* bsq = (float*)(ws + off); off += align256((size_t)N * sizeof(float));
  (void)ws_size; (void)n_in; (void)out_size;

  quant_rows_kernel<<<M, 256, 0, stream>>>(x, scales, qx, ax, K, 0);
  quant_rows_kernel<<<N, 256, 0, stream>>>(weight, scales, qw, aw, K, 1);
  bias_fold_kernel<<<(N + 255) / 256, 256, 0, stream>>>(bias, scales, bsq, N);
  dim3 grid(N / 256, M / 128);
  sq_int8_gemm_kernel<<<grid, 256, 0, stream>>>(qx, qw, ax, aw, bsq,
                                                (float*)d_out, M, N, K);
}
